// GPRGNN_PRE_53901839565315
// MI455X (gfx1250) — compile-verified, matrix operand order checked by reading, not run on hardware
//
#include <hip/hip_runtime.h>
#include <hip/hip_bf16.h>

typedef __attribute__((ext_vector_type(16))) _Float16 v16h;
typedef __attribute__((ext_vector_type(8)))  _Float16 v8h;
typedef __attribute__((ext_vector_type(8)))  float    v8f;

#define GN_NODES   50000
#define GN_EDGES   800000
#define GN_FEAT    128
#define GN_HIDDEN  256
#define GN_CLASSES 40
#define GN_CPAD    48
#define GK_HOPS    10

// ---------------- degree / norm ----------------
__global__ void k_deg_init(float* deg, int n) {
    int i = blockIdx.x * blockDim.x + threadIdx.x;
    if (i < n) deg[i] = 1.0f;  // self-loop contribution
}

__global__ void k_deg_count(const int* col, float* deg, int E) {
    int e = blockIdx.x * blockDim.x + threadIdx.x;
    if (e < E)
        __hip_atomic_fetch_add(&deg[col[e]], 1.0f, __ATOMIC_RELAXED, __HIP_MEMORY_SCOPE_AGENT);
}

__global__ void k_dinv(const float* deg, float* dinv, int n) {
    int i = blockIdx.x * blockDim.x + threadIdx.x;
    if (i < n) {
        float d = deg[i];
        dinv[i] = (d > 0.0f) ? rsqrtf(d) : 0.0f;
    }
}

// h = x ; hidden = temp[0]*x
__global__ void k_init_h(const float* x, float* h, float* hidden, const float* temp, int n) {
    int i = blockIdx.x * blockDim.x + threadIdx.x;
    if (i < n) {
        float v = x[i];
        h[i] = v;
        hidden[i] = temp[0] * v;
    }
}

// h_next = dinv[i]^2 * h_cur   (self-loop edge; also zero-initializes h_next)
__global__ void k_selfloop(const float* dinv, const float* hc, float* hn, int nfeat_total) {
    int i = blockIdx.x * blockDim.x + threadIdx.x;
    if (i < nfeat_total) {
        int node = i >> 7;  // /128
        float d = dinv[node];
        hn[i] = d * d * hc[i];
    }
}

// one wave per edge: lane handles 4 consecutive features (float4), native f32 atomics
__global__ void k_scatter(const int* row, const int* col, const float* dinv,
                          const float* hc, float* hn, int E) {
    int wid  = (int)((blockIdx.x * blockDim.x + threadIdx.x) >> 5);
    int lane = threadIdx.x & 31;
    if (wid >= E) return;
    int r = row[wid];
    int c = col[wid];
    float w = dinv[r] * dinv[c];
    float4 v = ((const float4*)(hc + (size_t)r * GN_FEAT))[lane];
    float* dst = hn + (size_t)c * GN_FEAT + lane * 4;
    __hip_atomic_fetch_add(dst + 0, w * v.x, __ATOMIC_RELAXED, __HIP_MEMORY_SCOPE_AGENT);
    __hip_atomic_fetch_add(dst + 1, w * v.y, __ATOMIC_RELAXED, __HIP_MEMORY_SCOPE_AGENT);
    __hip_atomic_fetch_add(dst + 2, w * v.z, __ATOMIC_RELAXED, __HIP_MEMORY_SCOPE_AGENT);
    __hip_atomic_fetch_add(dst + 3, w * v.w, __ATOMIC_RELAXED, __HIP_MEMORY_SCOPE_AGENT);
}

__global__ void k_accum(float* hidden, const float* hn, const float* temp, int k1, int n) {
    int i = blockIdx.x * blockDim.x + threadIdx.x;
    if (i < n) hidden[i] += temp[k1] * hn[i];
}

// ---------------- weight packing: f32 row-major -> f16 WMMA-B-lane order ----------------
// packed[kt][col][koff] = W[(kt*32+koff)*ncols_src + col], koff in 0..31
__global__ void k_pack_w1(const float* __restrict__ W1, _Float16* __restrict__ w1p) {
    int i = blockIdx.x * blockDim.x + threadIdx.x;     // 4*256*32 = 32768
    if (i >= 4 * GN_HIDDEN * 32) return;
    int koff = i & 31;
    int col  = (i >> 5) & (GN_HIDDEN - 1);
    int kt   = i >> 13;
    w1p[i] = (_Float16)W1[(size_t)(kt * 32 + koff) * GN_HIDDEN + col];
}

__global__ void k_pack_w2(const float* __restrict__ W2, _Float16* __restrict__ w2p) {
    int i = blockIdx.x * blockDim.x + threadIdx.x;     // 8*48*32 = 12288
    if (i >= 8 * GN_CPAD * 32) return;
    int koff = i & 31;
    int col  = (i >> 5) % GN_CPAD;
    int kt   = i / (GN_CPAD * 32);
    w2p[i] = (col < GN_CLASSES)
                 ? (_Float16)W2[(size_t)(kt * 32 + koff) * GN_CLASSES + col]
                 : (_Float16)0.0f;
}

// ---------------- GEMM1: z = relu(hidden[N,128] @ W1[128,256] + b1), z stored f16 ----------
// block (32,4): stage 16x128 A tile (f16) in LDS once; each wave computes 4 N-tiles.
// grid (3125, 1)
#define AS_STRIDE 136   // halves; 272B row stride -> lanes spread across LDS banks
__global__ void k_gemm1(const float* __restrict__ hidden, const _Float16* __restrict__ w1p,
                        const float* __restrict__ b1, _Float16* __restrict__ z) {
    __shared__ _Float16 As[16 * AS_STRIDE];

    int lane = threadIdx.x;            // 0..31
    int hi   = lane >> 4;
    int lm   = lane & 15;
    int mbase = blockIdx.x * 16;

    // ---- stage A: 128 threads, each converts 16 floats of one row ----
    {
        int t     = threadIdx.y * 32 + threadIdx.x;   // 0..127
        int row   = t >> 3;                           // 0..15
        int kchunk = (t & 7) * 16;                    // 0,16,...,112
        const float* src = hidden + (size_t)(mbase + row) * GN_FEAT + kchunk;
        float4 f0 = ((const float4*)src)[0];
        float4 f1 = ((const float4*)src)[1];
        float4 f2 = ((const float4*)src)[2];
        float4 f3 = ((const float4*)src)[3];
        v8h p0, p1;
        p0[0]=(_Float16)f0.x; p0[1]=(_Float16)f0.y; p0[2]=(_Float16)f0.z; p0[3]=(_Float16)f0.w;
        p0[4]=(_Float16)f1.x; p0[5]=(_Float16)f1.y; p0[6]=(_Float16)f1.z; p0[7]=(_Float16)f1.w;
        p1[0]=(_Float16)f2.x; p1[1]=(_Float16)f2.y; p1[2]=(_Float16)f2.z; p1[3]=(_Float16)f2.w;
        p1[4]=(_Float16)f3.x; p1[5]=(_Float16)f3.y; p1[6]=(_Float16)f3.z; p1[7]=(_Float16)f3.w;
        *(v8h*)&As[row * AS_STRIDE + kchunk]     = p0;
        *(v8h*)&As[row * AS_STRIDE + kchunk + 8] = p1;
    }
    __syncthreads();

    // ---- compute: 4 N-tiles per wave ----
    #pragma unroll
    for (int nt = 0; nt < 4; ++nt) {
        int ntile = nt * 4 + threadIdx.y;         // 0..15
        int ncol  = ntile * 16 + lm;              // 0..255
        v8f acc = {};

        #pragma unroll
        for (int kb = 0; kb < GN_FEAT; kb += 32) {
            v8h a0 = *(const v8h*)&As[lm * AS_STRIDE + kb + 8 * hi];
            v8h a1 = *(const v8h*)&As[lm * AS_STRIDE + kb + 16 + 8 * hi];
            v16h A;
            #pragma unroll
            for (int i = 0; i < 8; ++i) { A[i] = a0[i]; A[8 + i] = a1[i]; }

            const _Float16* bp = w1p + ((size_t)(kb >> 5) * GN_HIDDEN + ncol) * 32;
            v8h b0 = *(const v8h*)(bp + 8 * hi);
            v8h b1v = *(const v8h*)(bp + 16 + 8 * hi);
            v16h B;
            #pragma unroll
            for (int i = 0; i < 8; ++i) { B[i] = b0[i]; B[8 + i] = b1v[i]; }

            acc = __builtin_amdgcn_wmma_f32_16x16x32_f16(false, A, false, B, (short)0, acc, false, false);
        }

        float bias = b1[ncol];
        #pragma unroll
        for (int v = 0; v < 8; ++v) {
            float val = acc[v] + bias;
            val = val > 0.0f ? val : 0.0f;        // relu
            z[(size_t)(mbase + v + 8 * hi) * GN_HIDDEN + ncol] = (_Float16)val;
        }
    }
}

// ---------------- GEMM2: out = z[N,256] @ W2[256,40] + b2 ; N padded to 48 ----------------
// block 32 (one wave, one 16x16 tile); grid (3125, 3)
__global__ void k_gemm2(const _Float16* __restrict__ z, const _Float16* __restrict__ w2p,
                        const float* __restrict__ b2, float* __restrict__ out) {
    int lane = threadIdx.x;
    int hi   = lane >> 4;
    int lm   = lane & 15;
    int mbase = blockIdx.x * 16;
    int ncol  = blockIdx.y * 16 + lm;           // 0..47

    const _Float16* arow = z + (size_t)(mbase + lm) * GN_HIDDEN;
    v8f acc = {};

    #pragma unroll
    for (int kb = 0; kb < GN_HIDDEN; kb += 32) {
        v8h a0 = *(const v8h*)(arow + kb + 8 * hi);
        v8h a1 = *(const v8h*)(arow + kb + 16 + 8 * hi);
        v16h A;
        #pragma unroll
        for (int i = 0; i < 8; ++i) { A[i] = a0[i]; A[8 + i] = a1[i]; }

        const _Float16* bp = w2p + ((size_t)(kb >> 5) * GN_CPAD + ncol) * 32;
        v8h b0 = *(const v8h*)(bp + 8 * hi);
        v8h b1v = *(const v8h*)(bp + 16 + 8 * hi);
        v16h B;
        #pragma unroll
        for (int i = 0; i < 8; ++i) { B[i] = b0[i]; B[8 + i] = b1v[i]; }

        acc = __builtin_amdgcn_wmma_f32_16x16x32_f16(false, A, false, B, (short)0, acc, false, false);
    }

    if (ncol < GN_CLASSES) {
        float bias = b2[ncol];
        #pragma unroll
        for (int v = 0; v < 8; ++v)
            out[(size_t)(mbase + v + 8 * hi) * GN_CLASSES + ncol] = acc[v] + bias;
    }
}

// ---------------- log-softmax over 40 classes, in-place on d_out ----------------
__global__ void k_logsoftmax(float* out, int n) {
    int i = blockIdx.x * blockDim.x + threadIdx.x;
    if (i >= n) return;
    float* row = out + (size_t)i * GN_CLASSES;
    float m = -3.4e38f;
    #pragma unroll
    for (int j = 0; j < GN_CLASSES; ++j) m = fmaxf(m, row[j]);
    float s = 0.0f;
    #pragma unroll
    for (int j = 0; j < GN_CLASSES; ++j) s += __expf(row[j] - m);
    float ls = __logf(s);
    #pragma unroll
    for (int j = 0; j < GN_CLASSES; ++j) row[j] = row[j] - m - ls;
}

extern "C" void kernel_launch(void* const* d_in, const int* in_sizes, int n_in,
                              void* d_out, int out_size, void* d_ws, size_t ws_size,
                              hipStream_t stream) {
    (void)in_sizes; (void)n_in; (void)out_size; (void)ws_size;

    const float* x    = (const float*)d_in[0];
    const int*   ei   = (const int*)d_in[1];     // [2, E] flat: row ptr = ei, col ptr = ei+E
    const float* temp = (const float*)d_in[2];
    const float* W1   = (const float*)d_in[3];
    const float* b1   = (const float*)d_in[4];
    const float* W2   = (const float*)d_in[5];
    const float* b2   = (const float*)d_in[6];
    float* out = (float*)d_out;

    const int N = GN_NODES, E = GN_EDGES;
    const int NF = N * GN_FEAT;

    float* deg    = (float*)d_ws;                 // N
    float* dinv   = deg + N;                      // N
    float* hA     = dinv + N;                     // N*128
    float* hB     = hA + NF;                      // N*128
    float* hidden = hB + NF;                      // N*128
    _Float16* z   = (_Float16*)(hidden + NF);     // N*256 f16
    _Float16* w1p = z + (size_t)N * GN_HIDDEN;    // 4*256*32 f16
    _Float16* w2p = w1p + 4 * GN_HIDDEN * 32;     // 8*48*32 f16

    const int* erow = ei;
    const int* ecol = ei + E;

    // normalization
    k_deg_init  <<<(N + 255) / 256, 256, 0, stream>>>(deg, N);
    k_deg_count <<<(E + 255) / 256, 256, 0, stream>>>(ecol, deg, E);
    k_dinv      <<<(N + 255) / 256, 256, 0, stream>>>(deg, dinv, N);

    // weight packing (one-time, overlaps nothing important)
    k_pack_w1   <<<(4 * GN_HIDDEN * 32 + 255) / 256, 256, 0, stream>>>(W1, w1p);
    k_pack_w2   <<<(8 * GN_CPAD * 32 + 255) / 256, 256, 0, stream>>>(W2, w2p);

    // h = x ; hidden = temp[0]*x
    k_init_h    <<<(NF + 255) / 256, 256, 0, stream>>>(x, hA, hidden, temp, NF);

    // K hops: ping-pong hA/hB (fixed deterministic schedule)
    float* hc = hA;
    float* hn = hB;
    for (int k = 0; k < GK_HOPS; ++k) {
        k_selfloop <<<(NF + 255) / 256, 256, 0, stream>>>(dinv, hc, hn, NF);
        k_scatter  <<<(E * 32 + 255) / 256, 256, 0, stream>>>(erow, ecol, dinv, hc, hn, E);
        k_accum    <<<(NF + 255) / 256, 256, 0, stream>>>(hidden, hn, temp, k + 1, NF);
        float* t = hc; hc = hn; hn = t;
    }

    // MLP via WMMA
    {
        dim3 grid(GN_NODES / 16, 1);   // 3125 blocks; each does all 16 N-tiles
        dim3 block(32, 4);
        k_gemm1<<<grid, block, 0, stream>>>(hidden, w1p, b1, z);
    }
    {
        dim3 grid(GN_NODES / 16, 3);   // N padded 40 -> 48
        dim3 block(32, 1);
        k_gemm2<<<grid, block, 0, stream>>>(z, w2p, b2, out);
    }
    k_logsoftmax<<<(N + 255) / 256, 256, 0, stream>>>(out, N);
}
